// ResAttnConv2d_80204219285685
// MI455X (gfx1250) — compile-verified, matrix operand order checked by reading, not run on hardware
//
#include <hip/hip_runtime.h>
#include <hip/hip_bf16.h>
#include <stdint.h>

typedef __attribute__((ext_vector_type(16))) _Float16 v16h;
typedef __attribute__((ext_vector_type(8)))  _Float16 v8h;
typedef __attribute__((ext_vector_type(8)))  float    v8f;
typedef __attribute__((ext_vector_type(4)))  float    v4f;

#define NB   32            // batch
#define HW_  4096          // 64*64
#define CH_  32            // half channels
#define CO_  64            // out channels
#define BHW  (NB * HW_)    // 131072
#define KDIM 288           // 9 taps * 32 channels, tap-major: k = t*32 + c
#define PP   448           // padded 21*21 phase-plane row (441 -> 448)

#define CAT16(lo, hi) __builtin_shufflevector(lo, hi, 0, 1, 2, 3, 4, 5, 6, 7, \
                                              8, 9, 10, 11, 12, 13, 14, 15)

// ---- CDNA5 async global->LDS via inline asm (ASYNCcnt-tracked) ------------
__device__ __forceinline__ void async_copy_b128(_Float16* dst_lds, const _Float16* src_glb) {
#if defined(__gfx1250__)
  uint32_t ldsoff = (uint32_t)(uintptr_t)dst_lds;   // low 32 bits = LDS byte offset
  uint64_t gaddr  = (uint64_t)(uintptr_t)src_glb;
  asm volatile("global_load_async_to_lds_b128 %0, %1, off"
               :: "v"(ldsoff), "v"(gaddr) : "memory");
#else
  *(v8h*)dst_lds = *(const v8h*)src_glb;
#endif
}
__device__ __forceinline__ void wait_async() {
#if defined(__gfx1250__)
#if __has_builtin(__builtin_amdgcn_s_wait_asynccnt)
  __builtin_amdgcn_s_wait_asynccnt(0);
#else
  asm volatile("s_wait_asynccnt 0x0" ::: "memory");
#endif
#endif
}

// ---------------- BatchNorm statistics: per-channel mean / rstd ------------
__global__ void bn_stats_kernel(const float* __restrict__ src,
                                float* __restrict__ mean,
                                float* __restrict__ rstd) {
  __shared__ float ssum[256], ssq[256];
  int c = blockIdx.x, tid = threadIdx.x;
  float s = 0.f, q = 0.f;
  for (int i = tid; i < BHW / 4; i += 256) {
    int b = i >> 10, l4 = i & 1023;
    v4f v = *(const v4f*)(src + (size_t)b * CO_ * HW_ + (size_t)c * HW_ + l4 * 4);
#pragma unroll
    for (int e = 0; e < 4; ++e) { s += v[e]; q += v[e] * v[e]; }
  }
  ssum[tid] = s; ssq[tid] = q; __syncthreads();
  for (int st = 128; st > 0; st >>= 1) {
    if (tid < st) { ssum[tid] += ssum[tid + st]; ssq[tid] += ssq[tid + st]; }
    __syncthreads();
  }
  if (tid == 0) {
    float m = ssum[0] / (float)BHW;
    float v = ssq[0] / (float)BHW - m * m;
    mean[c] = m;
    rstd[c] = rsqrtf(v + 1e-5f);
  }
}

// ---------------- weight f32 -> f16 with tap-major repack ------------------
__global__ void cvt_repack_kernel(const float* __restrict__ s,
                                  _Float16* __restrict__ d, int n) {
  int i = blockIdx.x * 256 + threadIdx.x;
  if (i >= n) return;
  int oc = i / KDIM, k = i % KDIM;
  int t = k >> 5, c = k & 31;
  d[i] = (_Float16)s[((size_t)oc * CH_ + c) * 9 + t];
}

// ---------------- zero the 441..447 pad of the phase-plane buffers ---------
__global__ void zero_pad_kernel(_Float16* __restrict__ y1r,
                                _Float16* __restrict__ y2r) {
  int idx = blockIdx.x * 256 + threadIdx.x;
  v8h z = {};
  if (idx < CH_ * NB * 9) *(v8h*)(y1r + (size_t)idx * PP + 440) = z;
  if (idx < CO_ * NB * 9) *(v8h*)(y2r + (size_t)idx * PP + 440) = z;
}

// ---------------- h = 0.5*bn(x_half) + 0.01*a_pre*x_half  (to f16) ---------
__global__ void make_h_kernel(const float* __restrict__ x,
                              const float* __restrict__ mean,
                              const float* __restrict__ rstd,
                              const float* __restrict__ a_pre, int branch,
                              _Float16* __restrict__ h) {
  int i = blockIdx.x * 256 + threadIdx.x;     // over CH_*BHW/8
  if (i >= CH_ * BHW / 8) return;
  int c0 = (i & 3) * 8, rest = i >> 2;        // rest = b*4096 + l
  int l = rest & (HW_ - 1), b = rest >> 12;
  float ap = 0.01f * a_pre[branch];
  const float* xb = x + (size_t)b * CO_ * HW_ + (size_t)((branch & 1) * CH_) * HW_ + l;
  v8h outv;
#pragma unroll
  for (int e = 0; e < 8; ++e) {
    int g = (branch & 1) * CH_ + c0 + e;
    float r = rstd[g], m = mean[g];
    outv[e] = (_Float16)(xb[(size_t)(c0 + e) * HW_] * (0.5f * r + ap) - 0.5f * m * r);
  }
  *(v8h*)(h + (size_t)rest * CH_ + c0) = outv;
}

// ---------------- Implicit-GEMM 3x3 conv via WMMA --------------------------
// MODE: 0 = f16 [o][b][hw], 1 = f16 [b][hw][c], 2 = f32 [b][o][hw],
//       3 = f16 phase-plane [o][b][9][448] (attn3 operand layout)
template <int MODE>
__global__ void conv3_wmma_kernel(const _Float16* __restrict__ A, long a_bstride,
                                  const _Float16* __restrict__ in,
                                  _Float16* __restrict__ outh,
                                  float* __restrict__ outf) {
  __shared__ _Float16 sh_t[3 * 66 * 40];
  int row = blockIdx.y;
  int b = row >> 6, y = row & 63;
  int tid = threadIdx.x;
  const _Float16* hbase = in + (size_t)b * HW_ * CH_;
  for (int id = tid; id < 792; id += 256) {
    int chunk = id & 3, sp = id >> 2;      // sp = r*66 + xx
    int xx = sp % 66, r = sp / 66;
    int gx = xx - 1, gy = y + r - 1;
    int off = sp * 40 + chunk * 8;
    if (gx >= 0 && gx < 64 && gy >= 0 && gy < 64) {
      async_copy_b128(sh_t + off, hbase + (size_t)(gy * 64 + gx) * CH_ + chunk * 8);
    } else {
      v8h z = {};
      *(v8h*)(sh_t + off) = z;
    }
  }
  wait_async();
  __syncthreads();

  int w = tid >> 5, lane = tid & 31;
  int mi = w >> 2, ni = w & 3;
  int hi = (lane >> 4) & 1, lanelo = lane & 15;
  int co0 = blockIdx.x * 32 + mi * 16;
  int xcol = ni * 16 + lanelo;
  const _Float16* Arow0 = A + (size_t)b * a_bstride + (size_t)(co0 + lanelo) * KDIM + 8 * hi;

  v8f acc = {};
#pragma unroll
  for (int t = 0; t < 9; ++t) {            // one tap per 32-wide K step
    int kh = t / 3, kw = t % 3;
    const _Float16* Ar = Arow0 + t * 32;
    v8h alo = *(const v8h*)(Ar);
    v8h ahi = *(const v8h*)(Ar + 16);
    const _Float16* Br = &sh_t[(kh * 66 + xcol + kw) * 40 + 8 * hi];
    v8h blo = *(const v8h*)(Br);
    v8h bhi = *(const v8h*)(Br + 16);
    acc = __builtin_amdgcn_wmma_f32_16x16x32_f16(false, CAT16(alo, ahi),
                                                 false, CAT16(blo, bhi),
                                                 (short)0, acc, false, false);
  }
  int xout = ni * 16 + lanelo;
  int p = 0, ly = 0, lx = 0;
  if (MODE == 3) {
    p = (y % 3) * 3 + xout % 3;
    ly = y / 3; lx = xout / 3;
  }
#pragma unroll
  for (int r2 = 0; r2 < 8; ++r2) {
    int o = co0 + r2 + 8 * hi;
    float v = acc[r2];
    if (MODE == 0)
      outh[(size_t)o * BHW + (size_t)b * HW_ + y * 64 + xout] = (_Float16)v;
    else if (MODE == 1)
      outh[((size_t)b * HW_ + y * 64 + xout) * CH_ + o] = (_Float16)v;
    else if (MODE == 2)
      outf[(size_t)b * CO_ * HW_ + (size_t)o * HW_ + y * 64 + xout] = v;
    else {
      if (ly < 21 && lx < 21)
        outh[((size_t)(o * NB + b) * 9 + p) * PP + ly * 21 + lx] = (_Float16)v;
    }
  }
}

// ---------------- k=3 attention einsum (phase-plane inputs) ----------------
__global__ void attn3_wmma_kernel(const _Float16* __restrict__ y1r,
                                  const _Float16* __restrict__ y2r,
                                  const float* __restrict__ aw,
                                  float* __restrict__ akp) {
  int bt = blockIdx.x;
  int b = bt / 9, t = bt % 9;
  int tid = threadIdx.x;
  int w = tid >> 5, lane = tid & 31;
  int mi = w >> 1, ni = w & 1;
  int hi = (lane >> 4) & 1, lanelo = lane & 15;
  int o = mi * 16 + lanelo;
  int ii = ni * 16 + lanelo;
  const _Float16* Arow = y2r + ((size_t)(o * NB + b) * 9 + t) * PP + 8 * hi;
  const _Float16* Brow = y1r + ((size_t)(ii * NB + b) * 9 + t) * PP + 8 * hi;
  v8f acc = {};
#pragma unroll
  for (int ks = 0; ks < 14; ++ks) {        // 448 = 14 * 32
    int base = ks * 32;
    v8h alo = *(const v8h*)(Arow + base);
    v8h ahi = *(const v8h*)(Arow + base + 16);
    v8h blo = *(const v8h*)(Brow + base);
    v8h bhi = *(const v8h*)(Brow + base + 16);
    acc = __builtin_amdgcn_wmma_f32_16x16x32_f16(false, CAT16(alo, ahi),
                                                 false, CAT16(blo, bhi),
                                                 (short)0, acc, false, false);
  }
  const float inv = 0.05892556509887896f;  // 1/sqrt(32*9)
#pragma unroll
  for (int r = 0; r < 8; ++r) {
    int oo = mi * 16 + r + 8 * hi;
    size_t idx = ((size_t)(b * CO_ + oo) * CH_ + ii) * 9 + t;
    akp[idx] = acc[r] * inv + aw[(size_t)(oo * CH_ + ii) * 9 + t];
  }
}

// ---------------- k=1 attention einsum (pre-softmax) ----------------------
__global__ void attn1_wmma_kernel(const _Float16* __restrict__ z1c,
                                  const _Float16* __restrict__ z2c,
                                  float* __restrict__ spre) {
  int b = blockIdx.x;
  int tid = threadIdx.x;
  int w = tid >> 5, lane = tid & 31;
  int mi = w >> 1, ni = w & 1;
  int hi = (lane >> 4) & 1, lanelo = lane & 15;
  int o = mi * 16 + lanelo;
  int ii = ni * 16 + lanelo;
  const _Float16* Arow = z2c + (size_t)o * BHW + (size_t)b * HW_ + 8 * hi;
  const _Float16* Brow = z1c + (size_t)ii * BHW + (size_t)b * HW_ + 8 * hi;
  v8f acc = {};
#pragma unroll 4
  for (int ks = 0; ks < 128; ++ks) {       // K = 4096
    int base = ks * 32;
    v8h alo = *(const v8h*)(Arow + base);
    v8h ahi = *(const v8h*)(Arow + base + 16);
    v8h blo = *(const v8h*)(Brow + base);
    v8h bhi = *(const v8h*)(Brow + base + 16);
    acc = __builtin_amdgcn_wmma_f32_16x16x32_f16(false, CAT16(alo, ahi),
                                                 false, CAT16(blo, bhi),
                                                 (short)0, acc, false, false);
  }
  const float inv = 0.17677669529663687f;  // 1/sqrt(32)
#pragma unroll
  for (int r = 0; r < 8; ++r) {
    int oo = mi * 16 + r + 8 * hi;
    spre[((size_t)b * CO_ + oo) * CH_ + ii] = acc[r] * inv;
  }
}

// ---------------- softmax over cin=32 -------------------------------------
__global__ void softmax_kernel(const float* __restrict__ spre,
                               float* __restrict__ k2) {
  int idx = blockIdx.x * 256 + threadIdx.x;
  if (idx >= NB * CO_) return;
  const float* p = spre + (size_t)idx * CH_;
  float mx = -1e30f;
  for (int i = 0; i < CH_; ++i) mx = fmaxf(mx, p[i]);
  float e[CH_], s = 0.f;
  for (int i = 0; i < CH_; ++i) { e[i] = __expf(p[i] - mx); s += e[i]; }
  float r = 1.f / s;
  float* q = k2 + (size_t)idx * CH_;
  for (int i = 0; i < CH_; ++i) q[i] = e[i] * r;
}

// ---------------- ak = k2 * (k1 + aw) -> f16, tap-major K ------------------
__global__ void combine_ak_kernel(const float* __restrict__ akp,
                                  const float* __restrict__ k2,
                                  _Float16* __restrict__ akh) {
  int idx = blockIdx.x * 256 + threadIdx.x;
  if (idx >= NB * CO_ * KDIM) return;
  int bo = idx / KDIM, k = idx % KDIM;
  int t = k >> 5, i = k & 31;
  akh[idx] = (_Float16)(k2[(size_t)bo * CH_ + i] *
                        akp[(size_t)bo * KDIM + i * 9 + t]);
}

// ---------------- per-branch finalize + 4-branch combine (float4) ---------
__global__ void finalize_kernel(const float* __restrict__ v,
                                const float* __restrict__ vm,
                                const float* __restrict__ vr,
                                const float* __restrict__ a_post, int branch,
                                const float* __restrict__ x,
                                const float* __restrict__ cm,
                                float* __restrict__ yacc,
                                float* __restrict__ y2buf,
                                float* __restrict__ out) {
  int q = blockIdx.x * 256 + threadIdx.x;   // over N/4
  if (q >= NB * CO_ * HW_ / 4) return;
  size_t idx = (size_t)q * 4;
  int o = (int)((idx >> 12) & 63);
  float m = vm[o], r = vr[o], ap = 0.6f * a_post[branch];
  v4f vv = *(const v4f*)(v + idx);
  v4f term;
#pragma unroll
  for (int e = 0; e < 4; ++e) term[e] = 0.5f * (vv[e] - m) * r + ap * vv[e];
  if (branch == 0) {
    *(v4f*)(yacc + idx) = term;
  } else if (branch == 1) {
    v4f a = *(const v4f*)(yacc + idx);
#pragma unroll
    for (int e = 0; e < 4; ++e) a[e] += term[e];
    *(v4f*)(yacc + idx) = a;
  } else if (branch == 2) {
    *(v4f*)(y2buf + idx) = term;
  } else {
    float c0 = 0.99f * cm[0];
    v4f xa = *(const v4f*)(x + idx);
    v4f ya = *(const v4f*)(yacc + idx);
    v4f y2 = *(const v4f*)(y2buf + idx);
    v4f res;
#pragma unroll
    for (int e = 0; e < 4; ++e)
      res[e] = c0 * xa[e] + (ya[e] + y2[e] * term[e]) * 0.5773502691896258f;
    *(v4f*)(out + idx) = res;
  }
}

extern "C" void kernel_launch(void* const* d_in, const int* in_sizes, int n_in,
                              void* d_out, int out_size, void* d_ws, size_t ws_size,
                              hipStream_t stream) {
  const float* x      = (const float*)d_in[0];
  const float* wk1c1  = (const float*)d_in[1];
  const float* wk1c2  = (const float*)d_in[2];
  const float* wk2c1  = (const float*)d_in[3];
  const float* wk2c2  = (const float*)d_in[4];
  const float* wconv  = (const float*)d_in[5];
  const float* aw     = (const float*)d_in[6];
  const float* a_pre  = (const float*)d_in[7];
  const float* a_post = (const float*)d_in[8];
  const float* cm     = (const float*)d_in[9];
  float* out = (float*)d_out;

  char* ws = (char*)d_ws;
  size_t off = 0;
  auto alloc = [&](size_t bytes) -> void* {
    off = (off + 255) & ~(size_t)255;
    void* p = ws + off;
    off += bytes;
    return p;
  };

  float* xmean = (float*)alloc(64 * 4);
  float* xrstd = (float*)alloc(64 * 4);
  float* vmean = (float*)alloc(64 * 4);
  float* vrstd = (float*)alloc(64 * 4);

  const int W1N = 4 * 32 * KDIM, W2N = 4 * 64 * KDIM;
  _Float16* w1_16 = (_Float16*)alloc((size_t)W1N * 2);
  _Float16* w2_16 = (_Float16*)alloc((size_t)W2N * 2);
  _Float16* w3_16 = (_Float16*)alloc((size_t)W1N * 2);
  _Float16* w4_16 = (_Float16*)alloc((size_t)W2N * 2);
  _Float16* w5_16 = (_Float16*)alloc((size_t)W1N * 2);

  _Float16* hbuf  = (_Float16*)alloc((size_t)CH_ * BHW * 2);
  _Float16* y1r   = (_Float16*)alloc((size_t)CH_ * NB * 9 * PP * 2);
  _Float16* y2r   = (_Float16*)alloc((size_t)CO_ * NB * 9 * PP * 2);
  _Float16* z1c   = (_Float16*)alloc((size_t)CH_ * BHW * 2);
  _Float16* z2c   = (_Float16*)alloc((size_t)CO_ * BHW * 2);
  _Float16* uc    = (_Float16*)alloc((size_t)CH_ * BHW * 2);
  float*    akp   = (float*)alloc((size_t)NB * CO_ * KDIM * 4);
  float*    spre  = (float*)alloc((size_t)NB * CO_ * CH_ * 4);
  float*    k2b   = (float*)alloc((size_t)NB * CO_ * CH_ * 4);
  _Float16* akh   = (_Float16*)alloc((size_t)NB * CO_ * KDIM * 2);
  float*    vbuf  = (float*)alloc((size_t)NB * CO_ * HW_ * 4);
  float*    yaccb = (float*)alloc((size_t)NB * CO_ * HW_ * 4);
  float*    y2buf = (float*)alloc((size_t)NB * CO_ * HW_ * 4);

  dim3 blk(256);
  cvt_repack_kernel<<<(W1N + 255) / 256, blk, 0, stream>>>(wk1c1, w1_16, W1N);
  cvt_repack_kernel<<<(W2N + 255) / 256, blk, 0, stream>>>(wk1c2, w2_16, W2N);
  cvt_repack_kernel<<<(W1N + 255) / 256, blk, 0, stream>>>(wk2c1, w3_16, W1N);
  cvt_repack_kernel<<<(W2N + 255) / 256, blk, 0, stream>>>(wk2c2, w4_16, W2N);
  cvt_repack_kernel<<<(W1N + 255) / 256, blk, 0, stream>>>(wconv, w5_16, W1N);
  bn_stats_kernel<<<64, blk, 0, stream>>>(x, xmean, xrstd);
  zero_pad_kernel<<<(CO_ * NB * 9 + 255) / 256, blk, 0, stream>>>(y1r, y2r);

  const int nh8 = CH_ * BHW / 8;
  const int nmain4 = NB * CO_ * HW_ / 4;
  dim3 g32(1, NB * 64), g64(2, NB * 64);
  for (int br = 0; br < 4; ++br) {
    make_h_kernel<<<(nh8 + 255) / 256, blk, 0, stream>>>(x, xmean, xrstd, a_pre, br, hbuf);
    conv3_wmma_kernel<3><<<g32, blk, 0, stream>>>(w1_16 + (size_t)br * 32 * KDIM, 0, hbuf, y1r, nullptr);
    conv3_wmma_kernel<3><<<g64, blk, 0, stream>>>(w2_16 + (size_t)br * 64 * KDIM, 0, hbuf, y2r, nullptr);
    conv3_wmma_kernel<0><<<g32, blk, 0, stream>>>(w3_16 + (size_t)br * 32 * KDIM, 0, hbuf, z1c, nullptr);
    conv3_wmma_kernel<0><<<g64, blk, 0, stream>>>(w4_16 + (size_t)br * 64 * KDIM, 0, hbuf, z2c, nullptr);
    conv3_wmma_kernel<1><<<g32, blk, 0, stream>>>(w5_16 + (size_t)br * 32 * KDIM, 0, hbuf, uc, nullptr);
    attn3_wmma_kernel<<<NB * 9, blk, 0, stream>>>(y1r, y2r, aw + (size_t)br * 64 * 32 * 9, akp);
    attn1_wmma_kernel<<<NB, blk, 0, stream>>>(z1c, z2c, spre);
    softmax_kernel<<<(NB * CO_ + 255) / 256, blk, 0, stream>>>(spre, k2b);
    combine_ak_kernel<<<(NB * CO_ * KDIM + 255) / 256, blk, 0, stream>>>(akp, k2b, akh);
    conv3_wmma_kernel<2><<<g64, blk, 0, stream>>>(akh, (long)CO_ * KDIM, uc, nullptr, vbuf);
    bn_stats_kernel<<<64, blk, 0, stream>>>(vbuf, vmean, vrstd);
    finalize_kernel<<<(nmain4 + 255) / 256, blk, 0, stream>>>(vbuf, vmean, vrstd, a_post, br, x, cm, yaccb, y2buf, out);
  }
  (void)in_sizes; (void)n_in; (void)out_size; (void)ws_size;
}
